// StrokesSynthesis_80977313399301
// MI455X (gfx1250) — matrix operand order repaired and checked by reading, not compiled, not run on hardware
//
#include <hip/hip_runtime.h>

typedef _Float16 h16;
typedef _Float16 v16h __attribute__((ext_vector_type(16)));
typedef _Float16 v8h  __attribute__((ext_vector_type(8)));
typedef float    v8f  __attribute__((ext_vector_type(8)));

#define T_  512
#define B_  64
#define U_  64
#define H_  400
#define N_  10
#define C_  20
#define L_  60
#define IN_ 3
#define G_  1600
#define ASCALE 0.05f

// tile counts
#define KT_L0  15   // ceil(463/32)  layer0 K = 3+60+400  (layout x|w|h)
#define KT_L12 27   // ceil(863/32)  layer1/2 K = 400+400+60+3 (layout h|y0|w|x)
#define KT_W   13   // ceil(400/32)  window K
#define NT_W   2    // 30 -> 32 cols
#define KT_H   38   // ceil(1200/32) heads K
#define NT_H   8    // 121 -> 128 cols

#define A0_STRIDE 464   // layer0 LDS activation row stride (halves), 16B aligned rows

// output layout (flat float32, concatenated in reference return order)
#define OFF_MU  ((size_t)0)
#define OFF_LS  ((size_t)1310720)
#define OFF_RHO ((size_t)2621440)
#define OFF_PI  ((size_t)3276800)
#define OFF_EOS ((size_t)3932160)
#define OFF_H0H ((size_t)3964928)
#define OFF_H0C ((size_t)3990528)
#define OFF_H1H ((size_t)4016128)
#define OFF_H1C ((size_t)4041728)
#define OFF_H2H ((size_t)4067328)
#define OFF_H2C ((size_t)4092928)
#define OFF_PW  ((size_t)4118528)
#define OFF_PK  ((size_t)4122368)
#define OFF_AL  ((size_t)4123008)

__device__ __forceinline__ float sigm(float x) { return 1.0f / (1.0f + __expf(-x)); }

// 16-bit A-matrix (16x32) lane layout: row m = lane&15;
// halves 0..7  -> K = kt*32 + (lane&16?8:0) + h        (8 consecutive)
// halves 8..15 -> K = kt*32 + 16 + (lane&16?8:0) + h-8 (8 consecutive)
__device__ __forceinline__ int a_klocal(int lane, int h) {
    return h + ((lane & 16) ? 8 : 0) + ((h >= 8) ? 8 : 0);
}

__device__ __forceinline__ v16h cat8(v8h lo, v8h hi) {
    return __builtin_shufflevector(lo, hi, 0, 1, 2, 3, 4, 5, 6, 7, 8, 9, 10, 11, 12, 13, 14, 15);
}

// two 16-byte vector loads covering one A fragment for this lane
__device__ __forceinline__ v16h a_frag_vec(const h16* rowbase_k0, int lane) {
    const h16* p = rowbase_k0 + ((lane & 16) ? 8 : 0);
    return cat8(*(const v8h*)p, *(const v8h*)(p + 16));
}

__device__ __forceinline__ v8f wmma_f16(v16h a, v16h b, v8f c) {
    return __builtin_amdgcn_wmma_f32_16x16x32_f16(false, a, false, b, (short)0, c, false, false);
}

// ---------------- weight packing ----------------
// Packed B-fragment layout: halfIdx = ((nt*KT + kt)*32 + lane)*16 + h
// element = B[k][n] = Wcomb[n][k], k = kt*32 + (h&7) + (lane&16?8:0) + (h>=8?16:0), n = nt*16 + (lane&15)
// Wcomb[n][k] = k < Ka ? Wa[n*Ka+k] : (k < Ka+Kb ? Wb[n*Kb+(k-Ka)] : 0)
__global__ void pack_gate_w(const float* __restrict__ Wa, const float* __restrict__ Wb,
                            int Ka, int Kb, int Nreal, int KT, h16* __restrict__ dst, int total) {
    int idx = blockIdx.x * blockDim.x + threadIdx.x;
    if (idx >= total) return;
    int hh   = idx & 15;
    int lane = (idx >> 4) & 31;
    int frag = idx >> 9;
    int kt = frag % KT, nt = frag / KT;
    int n = nt * 16 + (lane & 15);
    int k = kt * 32 + (hh & 7) + ((lane & 16) ? 8 : 0) + ((hh >= 8) ? 16 : 0);
    float v = 0.f;
    if (n < Nreal) {
        if (k < Ka) v = Wa[n * Ka + k];
        else if (k < Ka + Kb) v = Wb[n * Kb + (k - Ka)];
    }
    dst[idx] = (h16)v;
}

__global__ void pack_head_w(const float* __restrict__ Wmu, const float* __restrict__ Wls,
                            const float* __restrict__ Wrho, const float* __restrict__ Wpi,
                            const float* __restrict__ Weos, h16* __restrict__ dst, int total) {
    int idx = blockIdx.x * blockDim.x + threadIdx.x;
    if (idx >= total) return;
    int hh   = idx & 15;
    int lane = (idx >> 4) & 31;
    int frag = idx >> 9;
    int kt = frag % KT_H, nt = frag / KT_H;
    int n = nt * 16 + (lane & 15);
    int k = kt * 32 + (hh & 7) + ((lane & 16) ? 8 : 0) + ((hh >= 8) ? 16 : 0);
    float v = 0.f;
    if (k < 1200) {
        if (n < 40)        v = Wmu[n * 1200 + k];
        else if (n < 80)   v = Wls[(n - 40) * 1200 + k];
        else if (n < 100)  v = Wrho[(n - 80) * 1200 + k];
        else if (n < 120)  v = Wpi[(n - 100) * 1200 + k];
        else if (n == 120) v = Weos[k];
    }
    dst[idx] = (h16)v;
}

__global__ void zero_f32(float* p, int n) {
    int i = blockIdx.x * blockDim.x + threadIdx.x;
    if (i < n) p[i] = 0.f;
}

// ---------------- layer 0 (LSTM + attention), persistent single workgroup ----------------
__global__ __launch_bounds__(1024) void layer0_kernel(
    const float* __restrict__ strokes, const int* __restrict__ text, const int* __restrict__ tlen,
    const float* __restrict__ h0h_in, const float* __restrict__ h0c_in,
    const float* __restrict__ prevw_in, const float* __restrict__ prevk_in,
    const float* __restrict__ b_ih0, const float* __restrict__ b_hh0,
    const float* __restrict__ b_win,
    const h16* __restrict__ wp0, const h16* __restrict__ wpw,
    float* __restrict__ gates, float* __restrict__ wacc,
    h16* __restrict__ y0h, h16* __restrict__ wth,
    float* __restrict__ out)
{
    // per-row k: [0..2]=x  [3..62]=w  [63..462]=h  [463]=0 pad; stride 464 (16B-aligned rows)
    __shared__ __align__(16) h16 Acat[B_ * A0_STRIDE + 64];
    __shared__ h16  owin[B_ * 20];    // window a,b (raw then exp'ed)
    __shared__ float klds[B_ * N_];   // window kappa (shared per step)

    const int tid  = threadIdx.x;
    const int lane = tid & 31;
    const int wave = tid >> 5;

    for (int e = tid; e < B_ * A0_STRIDE + 64; e += 1024) {
        if (e < B_ * A0_STRIDE) {
            int b = e / A0_STRIDE, k = e - A0_STRIDE * b;
            float v = 0.f;
            if (k < 3)        v = strokes[(size_t)b * IN_ + k];
            else if (k < 63)  v = prevw_in[b * L_ + (k - 3)];
            else if (k < 463) v = h0h_in[b * H_ + (k - 63)];
            Acat[e] = (h16)v;
        } else {
            Acat[e] = (h16)0.f;
        }
    }
    float c_reg[25];
#pragma unroll
    for (int i = 0; i < 25; ++i) {
        int e = i * 1024 + tid;
        int b = e / H_, j = e - H_ * b;
        c_reg[i] = h0c_in[b * H_ + j];
    }
    float k_reg = (tid < B_ * N_) ? prevk_in[tid] : 0.f;
    __syncthreads();

    for (int t = 0; t < T_; ++t) {
        // ---- gate GEMM: 4 m-tiles x 100 n-tiles, chunks of 4 n-tiles per wave ----
        for (int ch = wave; ch < 100; ch += 32) {
            int mt  = ch / 25;
            int nt0 = (ch % 25) * 4;
            v8f acc[4];
#pragma unroll
            for (int q = 0; q < 4; ++q)
#pragma unroll
                for (int v = 0; v < 8; ++v) acc[q][v] = 0.f;
            int m = mt * 16 + (lane & 15);
            v16h bf[4], bn[4];
#pragma unroll
            for (int q = 0; q < 4; ++q)
                bf[q] = *(const v16h*)(wp0 + ((size_t)((nt0 + q) * KT_L0) * 32 + lane) * 16);
            for (int kt = 0; kt < KT_L0; ++kt) {
                v16h af = a_frag_vec(&Acat[m * A0_STRIDE + kt * 32], lane);
                if (kt + 1 < KT_L0) {
#pragma unroll
                    for (int q = 0; q < 4; ++q)
                        bn[q] = *(const v16h*)(wp0 + ((size_t)((nt0 + q) * KT_L0 + kt + 1) * 32 + lane) * 16);
                }
#pragma unroll
                for (int q = 0; q < 4; ++q) acc[q] = wmma_f16(af, bf[q], acc[q]);
                if (kt + 1 < KT_L0) {
#pragma unroll
                    for (int q = 0; q < 4; ++q) bf[q] = bn[q];
                }
            }
            int mrow = mt * 16 + ((lane & 16) ? 8 : 0);
#pragma unroll
            for (int q = 0; q < 4; ++q) {
                int ncol = (nt0 + q) * 16 + (lane & 15);
#pragma unroll
                for (int v = 0; v < 8; ++v)
                    gates[(size_t)(mrow + v) * G_ + ncol] = acc[q][v];
            }
        }
        __threadfence(); __syncthreads();

        // ---- LSTM pointwise ----
#pragma unroll
        for (int i = 0; i < 25; ++i) {
            int e = i * 1024 + tid;
            int b = e / H_, j = e - H_ * b;
            float gi = gates[(size_t)b * G_ + j]        + b_ih0[j]        + b_hh0[j];
            float gf = gates[(size_t)b * G_ + 400 + j]  + b_ih0[400 + j]  + b_hh0[400 + j];
            float gg = gates[(size_t)b * G_ + 800 + j]  + b_ih0[800 + j]  + b_hh0[800 + j];
            float go = gates[(size_t)b * G_ + 1200 + j] + b_ih0[1200 + j] + b_hh0[1200 + j];
            float c = sigm(gf) * c_reg[i] + sigm(gi) * tanhf(gg);
            float h = sigm(go) * tanhf(c);
            c_reg[i] = c;
            Acat[b * A0_STRIDE + 63 + j] = (h16)h;
            y0h[((size_t)t * B_ + b) * H_ + j] = (h16)h;
            if (t == T_ - 1) { out[OFF_H0H + b * H_ + j] = h; out[OFF_H0C + b * H_ + j] = c; }
        }
        __syncthreads();

        // ---- window GEMM: h @ W_win^T  (4 m-tiles x 2 n-tiles, waves 0..7) ----
        if (wave < 8) {
            int mt = wave >> 1, nt = wave & 1;
            v8f acc;
#pragma unroll
            for (int v = 0; v < 8; ++v) acc[v] = 0.f;
            int m = mt * 16 + (lane & 15);
            for (int kt = 0; kt < KT_W; ++kt) {
                v16h af;
#pragma unroll
                for (int h = 0; h < 16; ++h) {
                    int k = kt * 32 + a_klocal(lane, h);
                    af[h] = (k < 400) ? Acat[m * A0_STRIDE + 63 + k] : (h16)0.f;
                }
                const v16h bf = *(const v16h*)(wpw + ((size_t)(nt * KT_W + kt) * 32 + lane) * 16);
                acc = wmma_f16(af, bf, acc);
            }
            int ncol = nt * 16 + (lane & 15);
            int mrow = mt * 16 + ((lane & 16) ? 8 : 0);
#pragma unroll
            for (int v = 0; v < 8; ++v) {
                int b = mrow + v;
                if (ncol < 20)      owin[b * 20 + ncol] = (h16)acc[v];
                else if (ncol < 30) klds[b * N_ + (ncol - 20)] = acc[v];
            }
        }
        __syncthreads();

        // ---- window params: a = exp(.), b = exp(.), kappa += exp(.)*scale ----
        if (tid < B_ * N_) {
            int b = tid / N_, n = tid - N_ * b;
            float a  = __expf((float)owin[b * 20 + n]      + b_win[n]);
            float bb = __expf((float)owin[b * 20 + 10 + n] + b_win[10 + n]);
            k_reg += __expf(klds[b * N_ + n] + b_win[20 + n]) * ASCALE;
            owin[b * 20 + n]      = (h16)a;
            owin[b * 20 + 10 + n] = (h16)bb;
            klds[b * N_ + n] = k_reg;
            if (t == T_ - 1) out[OFF_PK + b * N_ + n] = k_reg;
        }
        __syncthreads();

        // ---- phi + scatter into w ----
#pragma unroll
        for (int i = 0; i < 4; ++i) {
            int e = i * 1024 + tid;          // e = b*64+u
            int b = e >> 6, u = e & 63;
            float phi = 0.f;
            if (u < tlen[b]) {
                float uf = (float)u;
#pragma unroll
                for (int n = 0; n < N_; ++n) {
                    float d = klds[b * N_ + n] - uf;
                    phi += (float)owin[b * 20 + n] * __expf(-(float)owin[b * 20 + 10 + n] * d * d);
                }
                atomicAdd(&wacc[b * L_ + text[b * U_ + u]], phi);
            }
            out[OFF_AL + ((size_t)t * B_ + b) * U_ + u] = phi;
        }
        __threadfence(); __syncthreads();

        // ---- w readback + next x staging ----
#pragma unroll
        for (int i = 0; i < 4; ++i) {
            int e = i * 1024 + tid;
            if (e < B_ * L_) {
                int b = e / L_, l = e - L_ * b;
                float wv = wacc[e];
                wacc[e] = 0.f;
                Acat[b * A0_STRIDE + 3 + l] = (h16)wv;
                wth[((size_t)t * B_ + b) * L_ + l] = (h16)wv;
                if (t == T_ - 1) out[OFF_PW + e] = wv;
            } else if (e < B_ * L_ + B_ * IN_ && t + 1 < T_) {
                int xi = e - B_ * L_;
                int b = xi / IN_, ii = xi - IN_ * b;
                Acat[b * A0_STRIDE + ii] = (h16)strokes[((size_t)(t + 1) * B_ + b) * IN_ + ii];
            }
        }
        __threadfence(); __syncthreads();
    }
}

// ---------------- layers 1 / 2, persistent single workgroup ----------------
// combined K layout: [0..399]=h  [400..799]=y_prev  [800..859]=w  [860..862]=x  [863]=pad
__global__ __launch_bounds__(1024) void layer12_kernel(
    const float* __restrict__ strokes,
    const float* __restrict__ hh_in, const float* __restrict__ hc_in,
    const float* __restrict__ b_ih, const float* __restrict__ b_hh,
    const h16* __restrict__ wp, const h16* __restrict__ yprev, const h16* __restrict__ wth,
    float* __restrict__ gates, h16* __restrict__ yout,
    float* __restrict__ out, unsigned long long off_hh, unsigned long long off_hc)
{
    __shared__ __align__(16) h16 Hl[B_ * H_];   // rows 800B, 16B-aligned

    const int tid  = threadIdx.x;
    const int lane = tid & 31;
    const int wave = tid >> 5;

    for (int e = tid; e < B_ * H_; e += 1024) Hl[e] = (h16)hh_in[e];
    float c_reg[25];
#pragma unroll
    for (int i = 0; i < 25; ++i) c_reg[i] = hc_in[i * 1024 + tid];
    __syncthreads();

    for (int t = 0; t < T_; ++t) {
        size_t s0 = (size_t)t * B_;
        for (int ch = wave; ch < 100; ch += 32) {
            int mt  = ch / 25;
            int nt0 = (ch % 25) * 4;
            v8f acc[4];
#pragma unroll
            for (int q = 0; q < 4; ++q)
#pragma unroll
                for (int v = 0; v < 8; ++v) acc[q][v] = 0.f;
            int b = mt * 16 + (lane & 15);
            size_t s = s0 + b;
            v16h bf[4], bn[4];
#pragma unroll
            for (int q = 0; q < 4; ++q)
                bf[q] = *(const v16h*)(wp + ((size_t)((nt0 + q) * KT_L12) * 32 + lane) * 16);
            for (int kt = 0; kt < KT_L12; ++kt) {
                v16h af;
                if (kt <= 11) {
                    af = a_frag_vec(&Hl[b * H_ + kt * 32], lane);               // pure h (LDS)
                } else if (kt >= 13 && kt <= 24) {
                    af = a_frag_vec(yprev + s * H_ + (kt * 32 - 400), lane);    // pure y_prev
                } else {                                                         // boundary tiles
#pragma unroll
                    for (int h = 0; h < 16; ++h) {
                        int k = kt * 32 + a_klocal(lane, h);
                        h16 v;
                        if (k < 400)      v = Hl[b * H_ + k];
                        else if (k < 800) v = yprev[s * H_ + (k - 400)];
                        else if (k < 860) v = wth[s * L_ + (k - 800)];
                        else if (k < 863) v = (h16)strokes[s * IN_ + (k - 860)];
                        else              v = (h16)0.f;
                        af[h] = v;
                    }
                }
                if (kt + 1 < KT_L12) {
#pragma unroll
                    for (int q = 0; q < 4; ++q)
                        bn[q] = *(const v16h*)(wp + ((size_t)((nt0 + q) * KT_L12 + kt + 1) * 32 + lane) * 16);
                }
#pragma unroll
                for (int q = 0; q < 4; ++q) acc[q] = wmma_f16(af, bf[q], acc[q]);
                if (kt + 1 < KT_L12) {
#pragma unroll
                    for (int q = 0; q < 4; ++q) bf[q] = bn[q];
                }
            }
            int mrow = mt * 16 + ((lane & 16) ? 8 : 0);
#pragma unroll
            for (int q = 0; q < 4; ++q) {
                int ncol = (nt0 + q) * 16 + (lane & 15);
#pragma unroll
                for (int v = 0; v < 8; ++v)
                    gates[(size_t)(mrow + v) * G_ + ncol] = acc[q][v];
            }
        }
        __threadfence(); __syncthreads();

#pragma unroll
        for (int i = 0; i < 25; ++i) {
            int e = i * 1024 + tid;
            int b = e / H_, j = e - H_ * b;
            float gi = gates[(size_t)b * G_ + j]        + b_ih[j]        + b_hh[j];
            float gf = gates[(size_t)b * G_ + 400 + j]  + b_ih[400 + j]  + b_hh[400 + j];
            float gg = gates[(size_t)b * G_ + 800 + j]  + b_ih[800 + j]  + b_hh[800 + j];
            float go = gates[(size_t)b * G_ + 1200 + j] + b_ih[1200 + j] + b_hh[1200 + j];
            float c = sigm(gf) * c_reg[i] + sigm(gi) * tanhf(gg);
            float h = sigm(go) * tanhf(c);
            c_reg[i] = c;
            Hl[b * H_ + j] = (h16)h;
            yout[(s0 + b) * H_ + j] = (h16)h;
            if (t == T_ - 1) { out[off_hh + b * H_ + j] = h; out[off_hc + b * H_ + j] = c; }
        }
        __syncthreads();
    }
}

// ---------------- output heads: [32768 x 1216] @ [1216 x 128] ----------------
// K layout: [0..399]=y0  [400..799]=y1  [800..1199]=y2  [1200..1215]=pad
__global__ __launch_bounds__(256) void heads_kernel(
    const h16* __restrict__ wph,
    const h16* __restrict__ y0h, const h16* __restrict__ y1h, const h16* __restrict__ y2h,
    const float* __restrict__ b_mu, const float* __restrict__ b_ls, const float* __restrict__ b_rho,
    const float* __restrict__ b_pi, const float* __restrict__ b_eos,
    float* __restrict__ out)
{
    const int lane = threadIdx.x & 31;
    const int mt = blockIdx.x * 8 + (threadIdx.x >> 5);   // 2048 m-tiles

    v8f acc[NT_H];
#pragma unroll
    for (int q = 0; q < NT_H; ++q)
#pragma unroll
        for (int v = 0; v < 8; ++v) acc[q][v] = 0.f;

    size_t s = (size_t)mt * 16 + (lane & 15);
    for (int kt = 0; kt < KT_H; ++kt) {
        v16h af;
        if (kt <= 11) {
            af = a_frag_vec(y0h + s * H_ + kt * 32, lane);
        } else if (kt >= 13 && kt <= 24) {
            af = a_frag_vec(y1h + s * H_ + (kt * 32 - 400), lane);
        } else if (kt >= 25 && kt <= 36) {
            af = a_frag_vec(y2h + s * H_ + (kt * 32 - 800), lane);
        } else {
#pragma unroll
            for (int h = 0; h < 16; ++h) {
                int k = kt * 32 + a_klocal(lane, h);
                h16 v;
                if (k < 400)       v = y0h[s * H_ + k];
                else if (k < 800)  v = y1h[s * H_ + (k - 400)];
                else if (k < 1200) v = y2h[s * H_ + (k - 800)];
                else               v = (h16)0.f;
                af[h] = v;
            }
        }
#pragma unroll
        for (int q = 0; q < NT_H; ++q) {
            const v16h bf = *(const v16h*)(wph + ((size_t)(q * KT_H + kt) * 32 + lane) * 16);
            acc[q] = wmma_f16(af, bf, acc[q]);
        }
    }
    int mrow = mt * 16 + ((lane & 16) ? 8 : 0);
#pragma unroll
    for (int q = 0; q < NT_H; ++q) {
        int cc = q * 16 + (lane & 15);
#pragma unroll
        for (int v = 0; v < 8; ++v) {
            size_t sp = (size_t)(mrow + v);
            float val = acc[q][v];
            if (cc < 40)        out[OFF_MU  + sp * 40 + cc]        = val + b_mu[cc];
            else if (cc < 80)   out[OFF_LS  + sp * 40 + (cc - 40)] = val + b_ls[cc - 40];
            else if (cc < 100)  out[OFF_RHO + sp * 20 + (cc - 80)] = tanhf(val + b_rho[cc - 80]);
            else if (cc < 120)  out[OFF_PI  + sp * 20 + (cc - 100)] = val + b_pi[cc - 100];
            else if (cc == 120) out[OFF_EOS + sp]                   = val + b_eos[0];
        }
    }
}

// ---------------- host launcher ----------------
extern "C" void kernel_launch(void* const* d_in, const int* in_sizes, int n_in,
                              void* d_out, int out_size, void* d_ws, size_t ws_size,
                              hipStream_t stream) {
    (void)in_sizes; (void)n_in; (void)out_size; (void)ws_size;
    const float* strokes = (const float*)d_in[0];
    const int*   text    = (const int*)d_in[1];
    const int*   tlen    = (const int*)d_in[2];
    const float* h0h = (const float*)d_in[3];
    const float* h0c = (const float*)d_in[4];
    const float* h1h = (const float*)d_in[5];
    const float* h1c = (const float*)d_in[6];
    const float* h2h = (const float*)d_in[7];
    const float* h2c = (const float*)d_in[8];
    const float* prevw = (const float*)d_in[9];
    const float* prevk = (const float*)d_in[10];
    const float* W_ih0 = (const float*)d_in[11];
    const float* W_hh0 = (const float*)d_in[12];
    const float* b_ih0 = (const float*)d_in[13];
    const float* b_hh0 = (const float*)d_in[14];
    const float* W_ih1 = (const float*)d_in[15];
    const float* W_hh1 = (const float*)d_in[16];
    const float* b_ih1 = (const float*)d_in[17];
    const float* b_hh1 = (const float*)d_in[18];
    const float* W_ih2 = (const float*)d_in[19];
    const float* W_hh2 = (const float*)d_in[20];
    const float* b_ih2 = (const float*)d_in[21];
    const float* b_hh2 = (const float*)d_in[22];
    const float* W_win = (const float*)d_in[23];
    const float* b_win = (const float*)d_in[24];
    const float* W_mu  = (const float*)d_in[25];
    const float* b_mu  = (const float*)d_in[26];
    const float* W_ls  = (const float*)d_in[27];
    const float* b_ls  = (const float*)d_in[28];
    const float* W_rho = (const float*)d_in[29];
    const float* b_rho = (const float*)d_in[30];
    const float* W_pi  = (const float*)d_in[31];
    const float* b_pi  = (const float*)d_in[32];
    const float* W_eos = (const float*)d_in[33];
    const float* b_eos = (const float*)d_in[34];
    float* out = (float*)d_out;

    char* ws = (char*)d_ws;
    size_t off = 0;
    auto alloc = [&](size_t bytes) -> char* {
        char* p = ws + off;
        off += (bytes + 255) & ~(size_t)255;
        return p;
    };
    float* gates = (float*)alloc((size_t)B_ * G_ * 4);
    float* wacc  = (float*)alloc((size_t)B_ * L_ * 4);
    const size_t szWp0  = (size_t)100 * KT_L0 * 512;
    const size_t szWp12 = (size_t)100 * KT_L12 * 512;
    const size_t szWpw  = (size_t)NT_W * KT_W * 512;
    const size_t szWph  = (size_t)NT_H * KT_H * 512;
    const size_t szY    = (size_t)T_ * B_ * H_;
    const size_t szW    = (size_t)T_ * B_ * L_;
    h16* wp0 = (h16*)alloc(szWp0 * 2);
    h16* wp1 = (h16*)alloc(szWp12 * 2);
    h16* wp2 = (h16*)alloc(szWp12 * 2);
    h16* wpw = (h16*)alloc(szWpw * 2);
    h16* wph = (h16*)alloc(szWph * 2);
    h16* y0h = (h16*)alloc(szY * 2);
    h16* y1h = (h16*)alloc(szY * 2);
    h16* y2h = (h16*)alloc(szY * 2);
    h16* wth = (h16*)alloc(szW * 2);

    // weight packing (parallel). layer0: K = [x|w (63) | h (400)];
    // layers1/2: K = [h (400) | y0|w|x (463)]  -> swap Whh/Wih roles in packer.
    pack_gate_w<<<(int)((szWp0 + 255) / 256), 256, 0, stream>>>(W_ih0, W_hh0, 63, 400, G_, KT_L0, wp0, (int)szWp0);
    pack_gate_w<<<(int)((szWp12 + 255) / 256), 256, 0, stream>>>(W_hh1, W_ih1, 400, 463, G_, KT_L12, wp1, (int)szWp12);
    pack_gate_w<<<(int)((szWp12 + 255) / 256), 256, 0, stream>>>(W_hh2, W_ih2, 400, 463, G_, KT_L12, wp2, (int)szWp12);
    pack_gate_w<<<(int)((szWpw + 255) / 256), 256, 0, stream>>>(W_win, nullptr, 400, 0, 30, KT_W, wpw, (int)szWpw);
    pack_head_w<<<(int)((szWph + 255) / 256), 256, 0, stream>>>(W_mu, W_ls, W_rho, W_pi, W_eos, wph, (int)szWph);
    zero_f32<<<(B_ * L_ + 255) / 256, 256, 0, stream>>>(wacc, B_ * L_);

    // sequential recurrent layers (persistent workgroups)
    layer0_kernel<<<1, 1024, 0, stream>>>(strokes, text, tlen, h0h, h0c, prevw, prevk,
                                          b_ih0, b_hh0, b_win, wp0, wpw,
                                          gates, wacc, y0h, wth, out);
    layer12_kernel<<<1, 1024, 0, stream>>>(strokes, h1h, h1c, b_ih1, b_hh1, wp1, y0h, wth,
                                           gates, y1h, out, OFF_H1H, OFF_H1C);
    layer12_kernel<<<1, 1024, 0, stream>>>(strokes, h2h, h2c, b_ih2, b_hh2, wp2, y1h, wth,
                                           gates, y2h, out, OFF_H2H, OFF_H2C);

    // output heads (fully parallel WMMA GEMM)
    heads_kernel<<<256, 256, 0, stream>>>(wph, y0h, y1h, y2h, b_mu, b_ls, b_rho, b_pi, b_eos, out);
}
